// MultiTokenAttention_65171833749543
// MI455X (gfx1250) — matrix operand order repaired and checked by reading, not compiled, not run on hardware
//
#include <hip/hip_runtime.h>
#include <hip/hip_bf16.h>

// ---- problem constants (match the reference) ----
#define B_ 8
#define N_ 1024
#define C_ 768
#define H_ 12
#define D_ 64
#define G_ 6
#define QC 8              // q-rows per conv/softmax block
#define SROWS (QC + 2)    // rows cached in LDS (with halo)

// wave macro-tile: 4 M-tiles x 2 N-tiles of 16x16 => 64x32 output per wave
#define MT 4
#define NT 2

typedef _Float16 v16h __attribute__((ext_vector_type(16)));
typedef _Float16 v8h  __attribute__((ext_vector_type(8)));
typedef _Float16 v4h  __attribute__((ext_vector_type(4)));
typedef float    v8f  __attribute__((ext_vector_type(8)));

// ---------------------------------------------------------------------------
// fp32 -> fp16 conversion
// ---------------------------------------------------------------------------
__global__ __launch_bounds__(256)
void cvt_f32_f16_kernel(const float* __restrict__ x, _Float16* __restrict__ y, int n) {
  int i = (int)blockIdx.x * 256 + (int)threadIdx.x;
  if (i < n) y[i] = (_Float16)x[i];
}

// A fragment (16x32 f16): lane half lh holds K {lh*16..+15}
__device__ __forceinline__ v16h load_afrag(const _Float16* __restrict__ arow,
                                           int k0, int lh) {
  return *(const v16h*)(arow + k0 + lh * 16);
}

// ---------------------------------------------------------------------------
// Generic batched WMMA GEMM:  Out[m,n] = scale * sum_k A[m,k] * Bm[n,k] + bias[n]
// A: row-major [M,K] (lda), Bm: row-major [N,K] (ldb) -> computes A @ Bm^T.
// Batch bb decomposed as (bb/Hdiv, bb%Hdiv) with independent strides.
// One wave computes a 64x32 macro-tile: 8 v_wmma per K-step over 6 fragments.
// Requires M % 64 == 0, N % 32 == 0, K % 32 == 0 (all call sites satisfy this).
// ---------------------------------------------------------------------------
__global__ __launch_bounds__(256)
void wmma_gemm_kernel(const _Float16* __restrict__ A,
                      const _Float16* __restrict__ Bm,
                      const float* __restrict__ bias,
                      void* __restrict__ Out,
                      int M, int N, int K,
                      int lda, int ldb, int ldo,
                      long long bsA, long long hsA,
                      long long bsB, long long hsB,
                      long long bsO, long long hsO,
                      int Hdiv, float scale, int outF32) {
  const int bb  = (int)blockIdx.y;
  const int bo_ = bb / Hdiv, ho_ = bb % Hdiv;
  const long long offA = (long long)bo_ * bsA + (long long)ho_ * hsA;
  const long long offB = (long long)bo_ * bsB + (long long)ho_ * hsB;
  const long long offO = (long long)bo_ * bsO + (long long)ho_ * hsO;

  const int tilesN = N >> 5;                 // 32-wide macro columns
  const int tiles  = (M >> 6) * tilesN;      // 64-tall macro rows
  const int t = (int)blockIdx.x * 8 + ((int)threadIdx.x >> 5);
  if (t >= tiles) return;
  const int tm = t / tilesN, tn = t - tm * tilesN;
  const int lane = (int)threadIdx.x & 31;
  const int lh = lane >> 4, lm = lane & 15;

  const _Float16* ar[MT];
#pragma unroll
  for (int i = 0; i < MT; ++i)
    ar[i] = A + offA + (long long)(tm * 64 + i * 16 + lm) * lda;
  const _Float16* br[NT];
#pragma unroll
  for (int j = 0; j < NT; ++j)
    br[j] = Bm + offB + (long long)(tn * 32 + j * 16 + lm) * ldb + lh * 16;

  v8f acc[MT][NT];
#pragma unroll
  for (int i = 0; i < MT; ++i)
#pragma unroll
    for (int j = 0; j < NT; ++j) acc[i][j] = (v8f){};

  for (int k0 = 0; k0 < K; k0 += 32) {
    if (k0 + 256 < K) {                       // stream ~8 K-steps ahead
      __builtin_prefetch(ar[0] + k0 + 256, 0, 0);
      __builtin_prefetch(br[0] + k0 + 256, 0, 0);
    }
    v16h a[MT];
#pragma unroll
    for (int i = 0; i < MT; ++i) a[i] = load_afrag(ar[i], k0, lh);
    v16h b[NT];
#pragma unroll
    for (int j = 0; j < NT; ++j) b[j] = *(const v16h*)(br[j] + k0);
#pragma unroll
    for (int i = 0; i < MT; ++i)
#pragma unroll
      for (int j = 0; j < NT; ++j)
        acc[i][j] = __builtin_amdgcn_wmma_f32_16x16x32_f16(
            false, a[i], false, b[j], (short)0, acc[i][j], false, false);
  }

  // C/D layout: col = 16*j + (lane&15); row = 16*i + 8*(lane>>4) + vgpr
#pragma unroll
  for (int j = 0; j < NT; ++j) {
    const int col = tn * 32 + j * 16 + lm;
    const float bv = bias ? bias[col] : 0.0f;
#pragma unroll
    for (int i = 0; i < MT; ++i) {
#pragma unroll
      for (int v = 0; v < 8; ++v) {
        const int row = tm * 64 + i * 16 + lh * 8 + v;
        const float val = acc[i][j][v] * scale + bv;
        if (outF32) ((float*)Out)[offO + (long long)row * ldo + col] = val;
        else ((_Float16*)Out)[offO + (long long)row * ldo + col] = (_Float16)val;
      }
    }
  }
}

// ---------------------------------------------------------------------------
// V transpose: Vf flat [B*N, C] (f16, head h at cols h*64..h*64+63)
//           -> Vt [B,H,D,N] so PV GEMM B-operand is K-contiguous.
// ---------------------------------------------------------------------------
__global__ __launch_bounds__(256)
void vtrans_kernel(const _Float16* __restrict__ Vf, _Float16* __restrict__ Vt) {
  long long i = (long long)blockIdx.x * 256 + (int)threadIdx.x;
  int n = (int)(i & (N_ - 1));
  long long r = i >> 10;
  int d = (int)(r & (D_ - 1)); r >>= 6;
  int h = (int)(r % H_);
  int b = (int)(r / H_);
  Vt[i] = Vf[((long long)(b * N_ + n)) * C_ + h * D_ + d];
}

// ---------------------------------------------------------------------------
// Fused depthwise conv3x3 + row softmax + 2x2 cross-head mix.
// One block handles one (b, head-group) and QC consecutive q rows;
// both heads of the group are processed together so mixing is local.
// S is read exactly once (plus 2-row halo); mixed probs written once as f16.
// ---------------------------------------------------------------------------
__device__ __forceinline__ float conv3x3(const float* r, const float* w, int k) {
  float s = 0.0f;
#pragma unroll
  for (int dq = 0; dq < 3; ++dq) {
    const float* rp = r + dq * N_;
    float left  = (k > 0)      ? rp[k - 1] : 0.0f;
    float mid   = rp[k];
    float right = (k < N_ - 1) ? rp[k + 1] : 0.0f;
    s += w[dq * 3 + 0] * left + w[dq * 3 + 1] * mid + w[dq * 3 + 2] * right;
  }
  return s;
}

__global__ __launch_bounds__(256)
void conv_softmax_mix_kernel(const float* __restrict__ S,
                             const float* __restrict__ conv_w,
                             const float* __restrict__ head_mix,
                             _Float16* __restrict__ P) {
  extern __shared__ float smem[];   // [2][SROWS][N_]  (80 KB)
  __shared__ float red[16];
  const int tid  = (int)threadIdx.x;
  const int wave = tid >> 5;
  const int lane = tid & 31;
  const int q0 = (int)blockIdx.x * QC;
  const int g  = (int)blockIdx.y;
  const int b  = (int)blockIdx.z;
  const int h0 = g * 2, h1 = h0 + 1;

  float w0[9], w1[9];
#pragma unroll
  for (int i = 0; i < 9; ++i) {
    w0[i] = conv_w[h0 * 9 + i];
    w1[i] = conv_w[h1 * 9 + i];
  }
  // head_mix[g, h_in, k_out] flattened
  const float c00 = head_mix[g * 4 + 0];  // a0 -> out head0
  const float c10 = head_mix[g * 4 + 2];  // a1 -> out head0
  const float c01 = head_mix[g * 4 + 1];  // a0 -> out head1
  const float c11 = head_mix[g * 4 + 3];  // a1 -> out head1

  const long long sBase0 = (long long)(b * H_ + h0) * N_ * N_;
  const long long sBase1 = (long long)(b * H_ + h1) * N_ * N_;

  // cooperative load of SROWS rows (zero-padded halo) for both heads
  for (int idx = tid; idx < 2 * SROWS * N_; idx += 256) {
    int hh  = idx / (SROWS * N_);
    int rem = idx - hh * (SROWS * N_);
    int r = rem >> 10;
    int k = rem & (N_ - 1);
    int q = q0 - 1 + r;
    float v = 0.0f;
    if (q >= 0 && q < N_)
      v = S[(hh ? sBase1 : sBase0) + (long long)q * N_ + k];
    smem[idx] = v;
  }
  __syncthreads();

  for (int qi = 0; qi < QC; ++qi) {
    const int q = q0 + qi;
    float a0[4], a1[4];
#pragma unroll
    for (int j = 0; j < 4; ++j) {
      int k = tid * 4 + j;
      a0[j] = conv3x3(smem + qi * N_, w0, k);
      a1[j] = conv3x3(smem + SROWS * N_ + qi * N_, w1, k);
    }
    // ---- per-head row max ----
    float m0l = fmaxf(fmaxf(a0[0], a0[1]), fmaxf(a0[2], a0[3]));
    float m1l = fmaxf(fmaxf(a1[0], a1[1]), fmaxf(a1[2], a1[3]));
#pragma unroll
    for (int off = 16; off > 0; off >>= 1) {
      m0l = fmaxf(m0l, __shfl_xor(m0l, off, 32));
      m1l = fmaxf(m1l, __shfl_xor(m1l, off, 32));
    }
    __syncthreads();
    if (lane == 0) { red[wave] = m0l; red[8 + wave] = m1l; }
    __syncthreads();
    float m0 = red[0], m1 = red[8];
#pragma unroll
    for (int i = 1; i < 8; ++i) { m0 = fmaxf(m0, red[i]); m1 = fmaxf(m1, red[8 + i]); }

    // ---- exp + row sum ----
    float s0 = 0.0f, s1 = 0.0f;
#pragma unroll
    for (int j = 0; j < 4; ++j) {
      a0[j] = __expf(a0[j] - m0); s0 += a0[j];
      a1[j] = __expf(a1[j] - m1); s1 += a1[j];
    }
#pragma unroll
    for (int off = 16; off > 0; off >>= 1) {
      s0 += __shfl_xor(s0, off, 32);
      s1 += __shfl_xor(s1, off, 32);
    }
    __syncthreads();
    if (lane == 0) { red[wave] = s0; red[8 + wave] = s1; }
    __syncthreads();
    float t0 = 0.0f, t1 = 0.0f;
#pragma unroll
    for (int i = 0; i < 8; ++i) { t0 += red[i]; t1 += red[8 + i]; }
    const float r0 = 1.0f / t0, r1 = 1.0f / t1;

    // ---- cross-head mix + f16 store ----
    v4h o0, o1;
#pragma unroll
    for (int j = 0; j < 4; ++j) {
      float p0 = c00 * (a0[j] * r0) + c10 * (a1[j] * r1);
      float p1 = c01 * (a0[j] * r0) + c11 * (a1[j] * r1);
      o0[j] = (_Float16)p0;
      o1[j] = (_Float16)p1;
    }
    *(v4h*)(P + sBase0 + (long long)q * N_ + tid * 4) = o0;
    *(v4h*)(P + sBase1 + (long long)q * N_ + tid * 4) = o1;
  }
}

// ---------------------------------------------------------------------------
// host launcher
// ---------------------------------------------------------------------------
extern "C" void kernel_launch(void* const* d_in, const int* in_sizes, int n_in,
                              void* d_out, int out_size, void* d_ws, size_t ws_size,
                              hipStream_t stream) {
  (void)in_sizes; (void)n_in; (void)out_size; (void)ws_size;
  const float* query    = (const float*)d_in[0];
  const float* key      = (const float*)d_in[1];
  const float* value    = (const float*)d_in[2];
  const float* Wq       = (const float*)d_in[3];
  const float* bq       = (const float*)d_in[4];
  const float* Wk       = (const float*)d_in[5];
  const float* bk       = (const float*)d_in[6];
  const float* Wv       = (const float*)d_in[7];
  const float* bv       = (const float*)d_in[8];
  const float* conv_w   = (const float*)d_in[9];
  const float* head_mix = (const float*)d_in[10];
  const float* Wo       = (const float*)d_in[11];
  const float* bo       = (const float*)d_in[12];
  float* out = (float*)d_out;

  // ---- workspace carve-up ----
  char* ws = (char*)d_ws;
  unsigned long long off = 0;
  auto take = [&](unsigned long long bytes) -> char* {
    char* p = ws + off;
    off = (off + bytes + 255ull) & ~255ull;
    return p;
  };
  const unsigned long long BNC = (unsigned long long)B_ * N_ * C_;   // 6,291,456
  const unsigned long long CC  = (unsigned long long)C_ * C_;        // 589,824
  const unsigned long long SN  = (unsigned long long)B_ * H_ * N_ * N_;

  _Float16* q16  = (_Float16*)take(BNC * 2);
  _Float16* k16  = (_Float16*)take(BNC * 2);
  _Float16* v16  = (_Float16*)take(BNC * 2);
  _Float16* Wq16 = (_Float16*)take(CC * 2);
  _Float16* Wk16 = (_Float16*)take(CC * 2);
  _Float16* Wv16 = (_Float16*)take(CC * 2);
  _Float16* Wo16 = (_Float16*)take(CC * 2);
  _Float16* Qf   = (_Float16*)take(BNC * 2);   // projected Q, flat [B*N, C]
  _Float16* Kf   = (_Float16*)take(BNC * 2);
  _Float16* Vf   = (_Float16*)take(BNC * 2);
  _Float16* Vt   = (_Float16*)take(BNC * 2);   // [B,H,D,N]
  float*    Sb   = (float*)   take(SN * 4);    // scores
  _Float16* Pb   = (_Float16*)take(SN * 2);    // mixed probs
  _Float16* Oh   = (_Float16*)take(BNC * 2);   // attention out, flat [B*N, C]

  const dim3 blk(256);

  // ---- fp32 -> fp16 conversions ----
  cvt_f32_f16_kernel<<<(int)((BNC + 255) / 256), blk, 0, stream>>>(query, q16, (int)BNC);
  cvt_f32_f16_kernel<<<(int)((BNC + 255) / 256), blk, 0, stream>>>(key,   k16, (int)BNC);
  cvt_f32_f16_kernel<<<(int)((BNC + 255) / 256), blk, 0, stream>>>(value, v16, (int)BNC);
  cvt_f32_f16_kernel<<<(int)((CC + 255) / 256), blk, 0, stream>>>(Wq, Wq16, (int)CC);
  cvt_f32_f16_kernel<<<(int)((CC + 255) / 256), blk, 0, stream>>>(Wk, Wk16, (int)CC);
  cvt_f32_f16_kernel<<<(int)((CC + 255) / 256), blk, 0, stream>>>(Wv, Wv16, (int)CC);
  cvt_f32_f16_kernel<<<(int)((CC + 255) / 256), blk, 0, stream>>>(Wo, Wo16, (int)CC);

  // ---- Q/K/V projections: [8192,768] = x @ W^T + b ----
  const int tilesProj = (B_ * N_ / 64) * (C_ / 32);          // 128*24 = 3072 wave-tiles
  const dim3 gProj((tilesProj + 7) / 8, 1);
  wmma_gemm_kernel<<<gProj, blk, 0, stream>>>(q16, Wq16, bq, Qf,
      B_ * N_, C_, C_, C_, C_, C_, 0, 0, 0, 0, 0, 0, 1, 1.0f, 0);
  wmma_gemm_kernel<<<gProj, blk, 0, stream>>>(k16, Wk16, bk, Kf,
      B_ * N_, C_, C_, C_, C_, C_, 0, 0, 0, 0, 0, 0, 1, 1.0f, 0);
  wmma_gemm_kernel<<<gProj, blk, 0, stream>>>(v16, Wv16, bv, Vf,
      B_ * N_, C_, C_, C_, C_, C_, 0, 0, 0, 0, 0, 0, 1, 1.0f, 0);

  // ---- scores: per (b,h)  S = (Q_h @ K_h^T) / 8 ----
  {
    const int tilesS = (N_ / 64) * (N_ / 32);                // 16*32 = 512 wave-tiles
    const dim3 gS(tilesS / 8, B_ * H_);
    wmma_gemm_kernel<<<gS, blk, 0, stream>>>(Qf, Kf, nullptr, Sb,
        N_, N_, D_, C_, C_, N_,
        (long long)N_ * C_, (long long)D_,                    // A: Q head slice
        (long long)N_ * C_, (long long)D_,                    // B: K head slice
        (long long)H_ * N_ * N_, (long long)N_ * N_,          // out batch strides
        H_, 0.125f, 1);
  }

  // ---- fused conv3x3 + softmax + head mix -> f16 probs ----
  {
    const dim3 gC(N_ / QC, G_, B_);
    const size_t smemB = (size_t)2 * SROWS * N_ * sizeof(float);   // 80 KB
    conv_softmax_mix_kernel<<<gC, blk, smemB, stream>>>(Sb, conv_w, head_mix, Pb);
  }

  // ---- transpose V to [B,H,D,N] ----
  vtrans_kernel<<<(int)(BNC / 256), blk, 0, stream>>>(Vf, Vt);

  // ---- PV: per (b,h)  O = P @ V  (B-operand = Vt rows, K-contiguous) ----
  {
    const int tilesAV = (N_ / 64) * (D_ / 32);               // 16*2 = 32 wave-tiles
    const dim3 gAV(tilesAV / 8, B_ * H_);
    wmma_gemm_kernel<<<gAV, blk, 0, stream>>>(Pb, Vt, nullptr, Oh,
        N_, D_, N_, N_, N_, C_,
        (long long)H_ * N_ * N_, (long long)N_ * N_,          // A: P per head
        (long long)H_ * D_ * N_, (long long)D_ * N_,          // B: Vt per head
        (long long)N_ * C_, (long long)D_,                    // out: head cols of flat [B*N,C]
        H_, 1.0f, 0);
  }

  // ---- output projection: out = O @ Wo^T + bo (f32 out) ----
  wmma_gemm_kernel<<<gProj, blk, 0, stream>>>(Oh, Wo16, bo, out,
      B_ * N_, C_, C_, C_, C_, C_, 0, 0, 0, 0, 0, 0, 1, 1.0f, 1);
}
